// HopfieldCore_58609123721358
// MI455X (gfx1250) — compile-verified, matrix operand order checked by reading, not run on hardware
//
#include <hip/hip_runtime.h>
#include <hip/hip_bf16.h>

#define EDIM 512
#define NHEAD 16
#define HDIM 32
#define LQ 2048
#define SKLEN 2048
#define NBATCH 4
#define LN_EPS 1e-5f

typedef __attribute__((ext_vector_type(16))) __bf16 v16bf;
typedef __attribute__((ext_vector_type(8)))  float  v8f;

union BF16Vec {
  uint4 q[2];
  unsigned short s[16];
  v16bf v;
};

__device__ __forceinline__ unsigned short f2bf(float f) {
  union { float f; unsigned int u; } a; a.f = f;
  return (unsigned short)((a.u + 0x7FFFu + ((a.u >> 16) & 1u)) >> 16);
}

__device__ __forceinline__ void pack8(BF16Vec& v, int o, float4 lo, float4 hi) {
  v.s[o + 0] = f2bf(lo.x); v.s[o + 1] = f2bf(lo.y);
  v.s[o + 2] = f2bf(lo.z); v.s[o + 3] = f2bf(lo.w);
  v.s[o + 4] = f2bf(hi.x); v.s[o + 5] = f2bf(hi.y);
  v.s[o + 6] = f2bf(hi.z); v.s[o + 7] = f2bf(hi.w);
}

// ---------------------------------------------------------------------------
// One-shot fp32 -> bf16 (RNE) conversion, 8 elements/thread, b128 in/out.
// ---------------------------------------------------------------------------
__global__ __launch_bounds__(256) void cvt_f32_bf16(
    const float* __restrict__ in, unsigned short* __restrict__ out, int n8) {
  int t = blockIdx.x * blockDim.x + threadIdx.x;
  if (t >= n8) return;
  const float4* p = (const float4*)in + (size_t)t * 2;
  BF16Vec v;
  pack8(v, 0, p[0], p[1]);
  ((uint4*)out)[t] = v.q[0];
}

// ---------------------------------------------------------------------------
// Y[m, n] = sum_k Xb[m, k] * Wb[n, k] + bias[n]   (bf16 operands, f32 accum)
// Wave computes a 16x64 tile: one A fetch feeds 4 WMMAs (4 B col-groups).
// A (16x32) layout: lane = M&15, K-half = lane>>4; elems 0..7 = K base..+7,
// elems 8..15 = K base+16..+23.  B mirrors with N in place of M.
// C layout: lane holds col = lane&15, rows r + 8*(lane>>4).
// ---------------------------------------------------------------------------
__global__ __launch_bounds__(256) void gemm_bf16_wmma(
    const unsigned short* __restrict__ Xb, const unsigned short* __restrict__ Wb,
    const float* __restrict__ bias, float* __restrict__ Y,
    int M, int Ncols, int K) {
  const int lane = threadIdx.x & 31;
  const int wave = threadIdx.x >> 5;
  const int sub  = lane >> 4;
  const int idx  = lane & 15;
  const int m0 = blockIdx.y * 128 + wave * 16;
  const int n0 = blockIdx.x * 64;

  const unsigned short* xr = Xb + (size_t)(m0 + idx) * K + sub * 8;
  const unsigned short* wr = Wb + (size_t)(n0 + idx) * K + sub * 8;

  v8f acc0 = {}, acc1 = {}, acc2 = {}, acc3 = {};
  for (int k0 = 0; k0 < K; k0 += 32) {
    if (k0 + 32 < K) {
      __builtin_prefetch(xr + k0 + 32, 0, 1);
      __builtin_prefetch(wr + k0 + 32, 0, 1);
    }
    BF16Vec av;
    av.q[0] = *(const uint4*)(xr + k0);
    av.q[1] = *(const uint4*)(xr + k0 + 16);

    BF16Vec b0, b1, b2, b3;
    const unsigned short* w = wr + k0;
    b0.q[0] = *(const uint4*)w;             b0.q[1] = *(const uint4*)(w + 16);
    w += 16 * K;
    b1.q[0] = *(const uint4*)w;             b1.q[1] = *(const uint4*)(w + 16);
    w += 16 * K;
    b2.q[0] = *(const uint4*)w;             b2.q[1] = *(const uint4*)(w + 16);
    w += 16 * K;
    b3.q[0] = *(const uint4*)w;             b3.q[1] = *(const uint4*)(w + 16);

    acc0 = __builtin_amdgcn_wmma_f32_16x16x32_bf16(
        false, av.v, false, b0.v, (short)0, acc0, false, false);
    acc1 = __builtin_amdgcn_wmma_f32_16x16x32_bf16(
        false, av.v, false, b1.v, (short)0, acc1, false, false);
    acc2 = __builtin_amdgcn_wmma_f32_16x16x32_bf16(
        false, av.v, false, b2.v, (short)0, acc2, false, false);
    acc3 = __builtin_amdgcn_wmma_f32_16x16x32_bf16(
        false, av.v, false, b3.v, (short)0, acc3, false, false);
  }

  v8f* accs[4] = {&acc0, &acc1, &acc2, &acc3};
#pragma unroll
  for (int t = 0; t < 4; ++t) {
    const int col = n0 + t * 16 + idx;
    const float b = bias[col];
    v8f a = *accs[t];
#pragma unroll
    for (int r = 0; r < 8; ++r)
      Y[(size_t)(m0 + r + 8 * sub) * Ncols + col] = a[r] + b;
  }
}

// ---------------------------------------------------------------------------
// Reshape (len,NB,E) fp32 -> [NB, H, len, D] bf16.
// mode 0: q * beta[h]          mode 1: affine LayerNorm over D (k, v)
// ---------------------------------------------------------------------------
__global__ __launch_bounds__(256) void prep_heads(
    const float* __restrict__ x, const float* __restrict__ scaling,
    const float* __restrict__ lnw, const float* __restrict__ lnb,
    unsigned short* __restrict__ outh, int len, int mode) {
  int t = blockIdx.x * blockDim.x + threadIdx.x;
  int total = len * NBATCH * NHEAD;
  if (t >= total) return;
  int h   = t % NHEAD;
  int n   = (t / NHEAD) % NBATCH;
  int pos = t / (NHEAD * NBATCH);

  const float* xr = x + ((size_t)pos * NBATCH + n) * EDIM + h * HDIM;
  unsigned short* orow =
      outh + (((size_t)n * NHEAD + h) * (size_t)len + pos) * HDIM;

  float vals[HDIM];
  if (mode == 0) {
    const float beta = scaling[h];
#pragma unroll
    for (int d = 0; d < HDIM; ++d) vals[d] = xr[d] * beta;
  } else {
    float mu = 0.f;
#pragma unroll
    for (int d = 0; d < HDIM; ++d) mu += xr[d];
    mu *= (1.0f / HDIM);
    float var = 0.f;
#pragma unroll
    for (int d = 0; d < HDIM; ++d) {
      float c = xr[d] - mu;
      var += c * c;
    }
    var *= (1.0f / HDIM);
    const float rstd = rsqrtf(var + LN_EPS);
#pragma unroll
    for (int d = 0; d < HDIM; ++d)
      vals[d] = (xr[d] - mu) * rstd * lnw[d] + lnb[d];
  }
#pragma unroll
  for (int d = 0; d < HDIM; ++d) orow[d] = f2bf(vals[d]);
}

// ---------------------------------------------------------------------------
// Flash attention, one (n,h) per block; 4 waves x 16 query rows = 64 queries.
// K/V chunks staged memory->LDS with GLOBAL_LOAD_ASYNC_TO_LDS_B128 (ASYNCcnt),
// online softmax with 16-lane shfl_xor row reductions, P re-striped through
// LDS into WMMA A layout. Output written bf16 into (L, NB, E) for out-proj.
// ---------------------------------------------------------------------------
__global__ __launch_bounds__(128) void hopfield_attn(
    const unsigned short* __restrict__ Qh, const unsigned short* __restrict__ Kh,
    const unsigned short* __restrict__ Vh, unsigned short* __restrict__ attn_out) {
  __shared__ alignas(16) unsigned short Ks[32 * 32];
  __shared__ alignas(16) unsigned short Vs[32 * 32];
  __shared__ alignas(16) unsigned short Ps[4][16 * 32];

  const int lane = threadIdx.x & 31;
  const int wave = threadIdx.x >> 5;
  const int sub  = lane >> 4;
  const int idx  = lane & 15;
  const int qt = blockIdx.x * 64 + wave * 16;   // query tile base
  const int h  = blockIdx.y;
  const int n  = blockIdx.z;

  const size_t headQ = ((size_t)n * NHEAD + h) * (size_t)LQ * HDIM;
  const size_t headK = ((size_t)n * NHEAD + h) * (size_t)SKLEN * HDIM;

  // Q A-operand (beta already folded in), loaded once per wave.
  BF16Vec qa;
  {
    const unsigned short* qr = Qh + headQ + (size_t)(qt + idx) * HDIM + sub * 8;
    qa.q[0] = *(const uint4*)qr;
    qa.q[1] = *(const uint4*)(qr + 16);
  }

  // Per-thread staging addresses for the 32x32 bf16 K/V chunks.
  const int srow = threadIdx.x >> 2;
  const int scol = (threadIdx.x & 3) * 8;
  const unsigned ldsK = (unsigned)(unsigned long long)&Ks[srow * 32 + scol];
  const unsigned ldsV = (unsigned)(unsigned long long)&Vs[srow * 32 + scol];
  const unsigned short* gK = Kh + headK + (size_t)srow * HDIM + scol;
  const unsigned short* gV = Vh + headK + (size_t)srow * HDIM + scol;

  float m_run[8], l_run[8];
  v8f o0 = {}, o1 = {};
#pragma unroll
  for (int r = 0; r < 8; ++r) { m_run[r] = -1e30f; l_run[r] = 0.f; }
  const v8f zc = {};

  for (int s0 = 0; s0 < SKLEN; s0 += 32) {
    // Async memory -> LDS staging (no VGPR round trip), tracked by ASYNCcnt.
    {
      const unsigned short* kk = gK + (size_t)s0 * HDIM;
      const unsigned short* vv = gV + (size_t)s0 * HDIM;
      asm volatile("global_load_async_to_lds_b128 %0, %1, off"
                   :: "v"(ldsK), "v"(kk) : "memory");
      asm volatile("global_load_async_to_lds_b128 %0, %1, off"
                   :: "v"(ldsV), "v"(vv) : "memory");
      asm volatile("s_wait_asynccnt 0x0" ::: "memory");
    }
    __syncthreads();

    // scores (16 q x 32 keys) = Q(16x32) @ K^T : two 16x16x32 WMMAs
    v8f c0, c1;
    {
      BF16Vec kb;
      const unsigned short* kr = &Ks[idx * 32 + sub * 8];
      kb.q[0] = *(const uint4*)kr;
      kb.q[1] = *(const uint4*)(kr + 16);
      c0 = __builtin_amdgcn_wmma_f32_16x16x32_bf16(
          false, qa.v, false, kb.v, (short)0, zc, false, false);
      kr = &Ks[(16 + idx) * 32 + sub * 8];
      kb.q[0] = *(const uint4*)kr;
      kb.q[1] = *(const uint4*)(kr + 16);
      c1 = __builtin_amdgcn_wmma_f32_16x16x32_bf16(
          false, qa.v, false, kb.v, (short)0, zc, false, false);
    }

    // Online softmax: row max / rescale / exp / row sum.
    float mnew[8], scale[8];
#pragma unroll
    for (int r = 0; r < 8; ++r) {
      float v = fmaxf(c0[r], c1[r]);
#pragma unroll
      for (int off = 1; off < 16; off <<= 1)
        v = fmaxf(v, __shfl_xor(v, off, 32));
      mnew[r]  = fmaxf(m_run[r], v);
      scale[r] = __expf(m_run[r] - mnew[r]);
    }
    v8f p0, p1;
#pragma unroll
    for (int r = 0; r < 8; ++r) {
      p0[r] = __expf(c0[r] - mnew[r]);
      p1[r] = __expf(c1[r] - mnew[r]);
      float sum = p0[r] + p1[r];
#pragma unroll
      for (int off = 1; off < 16; off <<= 1)
        sum += __shfl_xor(sum, off, 32);
      l_run[r] = l_run[r] * scale[r] + sum;
      m_run[r] = mnew[r];
      o0[r] *= scale[r];
      o1[r] *= scale[r];
    }

    // Re-stripe P: C-layout (lane=col) -> A-layout (lane=row) via LDS.
    unsigned short* pw = &Ps[wave][0];
#pragma unroll
    for (int r = 0; r < 8; ++r) {
      const int row = r + 8 * sub;
      pw[row * 32 + idx]      = f2bf(p0[r]);
      pw[row * 32 + idx + 16] = f2bf(p1[r]);
    }
    // Same-wave LDS ops are in-order; no barrier needed for the RAW below.
    BF16Vec pa;
    {
      const unsigned short* pr = &pw[idx * 32 + sub * 8];
      pa.q[0] = *(const uint4*)pr;
      pa.q[1] = *(const uint4*)(pr + 16);
    }

    // V chunk as B (32 keys x 32 d), split into d 0..15 / 16..31.
    BF16Vec vb0, vb1;
#pragma unroll
    for (int j = 0; j < 8; ++j) {
      const int vr = sub * 8 + j;
      vb0.s[j]     = Vs[vr * 32 + idx];
      vb0.s[j + 8] = Vs[(vr + 16) * 32 + idx];
      vb1.s[j]     = Vs[vr * 32 + idx + 16];
      vb1.s[j + 8] = Vs[(vr + 16) * 32 + idx + 16];
    }
    o0 = __builtin_amdgcn_wmma_f32_16x16x32_bf16(
        false, pa.v, false, vb0.v, (short)0, o0, false, false);
    o1 = __builtin_amdgcn_wmma_f32_16x16x32_bf16(
        false, pa.v, false, vb1.v, (short)0, o1, false, false);

    __syncthreads();
  }

  // Normalize and scatter bf16 into (L, NB, E) for the out-projection GEMM.
#pragma unroll
  for (int r = 0; r < 8; ++r) {
    const int q     = qt + r + 8 * sub;
    const float inv = 1.0f / l_run[r];
    unsigned short* dst =
        attn_out + ((size_t)q * NBATCH + n) * EDIM + h * HDIM;
    dst[idx]      = f2bf(o0[r] * inv);
    dst[idx + 16] = f2bf(o1[r] * inv);
  }
}

// ---------------------------------------------------------------------------
extern "C" void kernel_launch(void* const* d_in, const int* in_sizes, int n_in,
                              void* d_out, int out_size, void* d_ws, size_t ws_size,
                              hipStream_t stream) {
  (void)in_sizes; (void)n_in; (void)out_size; (void)ws_size;

  const float* query   = (const float*)d_in[0];
  const float* key_in  = (const float*)d_in[1];
  const float* value   = (const float*)d_in[2];
  const float* scaling = (const float*)d_in[3];
  const float* in_w    = (const float*)d_in[4];
  const float* in_b    = (const float*)d_in[5];
  const float* pnw     = (const float*)d_in[6];
  const float* pnb     = (const float*)d_in[7];
  const float* out_w   = (const float*)d_in[8];
  const float* out_b   = (const float*)d_in[9];
  float* out = (float*)d_out;

  const size_t MB = 1u << 20;
  char* ws = (char*)d_ws;
  unsigned short* Xq = (unsigned short*)(ws + 0 * MB);    // (L*NB, E) bf16
  unsigned short* Xk = (unsigned short*)(ws + 8 * MB);
  unsigned short* Xv = (unsigned short*)(ws + 16 * MB);
  unsigned short* Wi = (unsigned short*)(ws + 24 * MB);   // (3E, E) bf16
  unsigned short* Wo = (unsigned short*)(ws + 26 * MB);   // (E, E) bf16
  float*          qp = (float*)(ws + 28 * MB);            // (L*NB, E) fp32
  float*          kp = (float*)(ws + 44 * MB);
  float*          vp = (float*)(ws + 60 * MB);
  unsigned short* Qh = (unsigned short*)(ws + 76 * MB);   // [NB,H,L,D] bf16
  unsigned short* Kh = (unsigned short*)(ws + 84 * MB);
  unsigned short* Vh = (unsigned short*)(ws + 92 * MB);
  unsigned short* Ah = (unsigned short*)(ws + 100 * MB);  // (L, NB, E) bf16

  const int M = LQ * NBATCH;                 // 8192 rows for all GEMMs

  // One-shot bf16 conversion of activations and weights.
  {
    const int nX8 = M * EDIM / 8;            // 524288
    cvt_f32_bf16<<<(nX8 + 255) / 256, 256, 0, stream>>>(query,  Xq, nX8);
    cvt_f32_bf16<<<(nX8 + 255) / 256, 256, 0, stream>>>(key_in, Xk, nX8);
    cvt_f32_bf16<<<(nX8 + 255) / 256, 256, 0, stream>>>(value,  Xv, nX8);
    const int nWi8 = 3 * EDIM * EDIM / 8;    // 98304
    cvt_f32_bf16<<<(nWi8 + 255) / 256, 256, 0, stream>>>(in_w, Wi, nWi8);
    const int nWo8 = EDIM * EDIM / 8;        // 32768
    cvt_f32_bf16<<<(nWo8 + 255) / 256, 256, 0, stream>>>(out_w, Wo, nWo8);
  }

  dim3 gblk(256);
  dim3 ggrid(EDIM / 64, M / 128);            // (8, 64)

  // In-projections: y = x @ W_slice^T + b_slice
  gemm_bf16_wmma<<<ggrid, gblk, 0, stream>>>(Xq, Wi,                 in_b,            qp, M, EDIM, EDIM);
  gemm_bf16_wmma<<<ggrid, gblk, 0, stream>>>(Xk, Wi + 1 * EDIM * EDIM, in_b + 1 * EDIM, kp, M, EDIM, EDIM);
  gemm_bf16_wmma<<<ggrid, gblk, 0, stream>>>(Xv, Wi + 2 * EDIM * EDIM, in_b + 2 * EDIM, vp, M, EDIM, EDIM);

  // Head reshape + beta (q) / LayerNorm (k, v) -> bf16
  const int total = LQ * NBATCH * NHEAD;     // 131072
  dim3 pblk(256), pgrid((total + 255) / 256);
  prep_heads<<<pgrid, pblk, 0, stream>>>(qp, scaling, pnw, pnb, Qh, LQ,    0);
  prep_heads<<<pgrid, pblk, 0, stream>>>(kp, scaling, pnw, pnb, Kh, SKLEN, 1);
  prep_heads<<<pgrid, pblk, 0, stream>>>(vp, scaling, pnw, pnb, Vh, SKLEN, 1);

  // Flash attention
  dim3 ablk(128);
  dim3 agrid(LQ / 64, NHEAD, NBATCH);        // (32, 16, 4)
  hopfield_attn<<<agrid, ablk, 0, stream>>>(Qh, Kh, Vh, Ah);

  // Out-projection straight into d_out
  gemm_bf16_wmma<<<ggrid, gblk, 0, stream>>>(Ah, Wo, out_b, out, M, EDIM, EDIM);
}
